// MultiRNN_60524679135325
// MI455X (gfx1250) — matrix operand order repaired and checked by reading, not compiled, not run on hardware
//
#include <hip/hip_runtime.h>
#include <hip/hip_bf16.h>

// ---------------------------------------------------------------------------
// MultiRNN on MI455X/gfx1250:
//   stage 1 (VALU):  h_t[c,:] = tanh(W[z[c]] @ E[x[c]] + U[z[c]] @ h[c])
//   stage 2 (WMMA):  logits[c,v] = dot(h_t[c,:], Vw[v,:]) + Vb[v]
// Stage 2 is store-BW bound (262 MB f32 logits @ 23.3 TB/s ~= 11 us).
// bf16 is converted ONCE at LDS staging; fragments are pure ds_load_b128
// matching the wave32 WMMA register layouts, so the inner loop is ds+wmma.
// ---------------------------------------------------------------------------

typedef __attribute__((ext_vector_type(16))) __bf16 v16bf;
typedef __attribute__((ext_vector_type(8)))  __bf16 v8bf;
typedef __attribute__((ext_vector_type(8)))  float  v8f;

#define BATCH 2048
#define HID   256
#define VOCAB 32000

__device__ __forceinline__ v8bf cvt8(float4 a, float4 b) {
    v8bf r;
    r[0] = (__bf16)a.x; r[1] = (__bf16)a.y; r[2] = (__bf16)a.z; r[3] = (__bf16)a.w;
    r[4] = (__bf16)b.x; r[5] = (__bf16)b.y; r[6] = (__bf16)b.z; r[7] = (__bf16)b.w;
    return r;
}

// ---------------------------------------------------------------------------
// Stage 1: one block per batch element, thread k computes output channel k.
// W/U banks (8 MB total) are L2-resident; e / h vectors staged in LDS.
// ---------------------------------------------------------------------------
__global__ __launch_bounds__(HID) void rnn_step_kernel(
        const int* __restrict__ x, const int* __restrict__ z,
        const float* __restrict__ h, const float* __restrict__ E,
        const float* __restrict__ W, const float* __restrict__ U,
        float* __restrict__ ht)
{
    __shared__ alignas(16) float se[HID];
    __shared__ alignas(16) float sh[HID];

    const int c = blockIdx.x;
    const int t = threadIdx.x;
    const int xc = x[c];
    const int zc = z[c];               // z shape (1, BATCH)

    se[t] = E[(size_t)xc * HID + t];
    sh[t] = h[(size_t)c  * HID + t];   // h shape (BATCH, 1, HID)
    __syncthreads();

    const float4* w4  = (const float4*)(W + ((size_t)zc * HID + t) * HID);
    const float4* u4  = (const float4*)(U + ((size_t)zc * HID + t) * HID);
    const float4* se4 = (const float4*)se;
    const float4* sh4 = (const float4*)sh;

    float acc = 0.f;
#pragma unroll 8
    for (int i = 0; i < HID / 4; ++i) {
        float4 w = w4[i], u = u4[i], e = se4[i], hv = sh4[i];
        acc += w.x * e.x + w.y * e.y + w.z * e.z + w.w * e.w;
        acc += u.x * hv.x + u.y * hv.y + u.z * hv.z + u.w * hv.w;
    }
    ht[(size_t)c * HID + t] = tanhf(acc);
}

// ---------------------------------------------------------------------------
// Stage 2: logits = Ht(2048x256) * Vw^T(256x32000) + Vb  via bf16 WMMA.
// 256 threads = 8 waves; WG tile 128(M) x 128(N), K-step 32, K = 256.
// Wave (mw,nw) in a 2x4 grid owns 64x32 = 4x2 tiles of 16x16 f32 accum.
// LDS holds bf16 (double-buffered); per-iteration pipeline:
//   global->regs(next) | wmma(cur) | cvt+store->LDS(next) | barrier
// ---------------------------------------------------------------------------
#define MT 128
#define NT 128
#define KSTEP 32
#define LSTR 40        // LDS row stride in bf16 (80 B: 16B-aligned, bank-spread)

__global__ __launch_bounds__(256) void logits_wmma_kernel(
        const float* __restrict__ Ht, const float* __restrict__ Vw,
        const float* __restrict__ Vb, float* __restrict__ logits)
{
    __shared__ alignas(16) __bf16 As[2][MT * LSTR];
    __shared__ alignas(16) __bf16 Bs[2][NT * LSTR];

    const int n0   = blockIdx.x * NT;
    const int m0   = blockIdx.y * MT;
    const int tid  = threadIdx.x;
    const int wave = tid >> 5;
    const int lane = tid & 31;
    const int mw   = wave >> 2;        // 0..1 -> M offset mw*64
    const int nw   = wave & 3;         // 0..3 -> N offset nw*32
    const int ll   = lane & 15;        // lane within half-wave (row/col id)
    const int hi   = lane >> 4;        // which K-half this lane holds

    // staging geometry: tile viewed as [128 rows][4 chunks of 8 floats]
    const int srow0 = (tid + 0)   >> 2, sc80 = ((tid + 0)   & 3) * 8;
    const int srow1 = (tid + 256) >> 2, sc81 = ((tid + 256) & 3) * 8;

    union Frag { v16bf v; v8bf h[2]; };

    v8f acc[4][2];
    const v8f zero8 = {0.f, 0.f, 0.f, 0.f, 0.f, 0.f, 0.f, 0.f};
#pragma unroll
    for (int i = 0; i < 4; ++i)
#pragma unroll
        for (int j = 0; j < 2; ++j) acc[i][j] = zero8;

    // ---- prologue: stage K-chunk 0 into buffer 0
#pragma unroll
    for (int r = 0; r < 2; ++r) {
        int row = r ? srow1 : srow0;
        int c8  = r ? sc81  : sc80;
        float4 a0 = *(const float4*)&Ht[(size_t)(m0 + row) * HID + c8];
        float4 a1 = *(const float4*)&Ht[(size_t)(m0 + row) * HID + c8 + 4];
        float4 b0 = *(const float4*)&Vw[(size_t)(n0 + row) * HID + c8];
        float4 b1 = *(const float4*)&Vw[(size_t)(n0 + row) * HID + c8 + 4];
        *(v8bf*)&As[0][row * LSTR + c8] = cvt8(a0, a1);
        *(v8bf*)&Bs[0][row * LSTR + c8] = cvt8(b0, b1);
    }
    __syncthreads();

#pragma unroll
    for (int kk = 0; kk < HID / KSTEP; ++kk) {
        const int cur = kk & 1;

        // ---- (1) issue next K-chunk global loads (latency hidden by WMMAs)
        float4 fa0[2], fa1[2], fb0[2], fb1[2];
        if (kk < HID / KSTEP - 1) {
            const int k0 = (kk + 1) * KSTEP;
#pragma unroll
            for (int r = 0; r < 2; ++r) {
                int row = r ? srow1 : srow0;
                int c8  = r ? sc81  : sc80;
                fa0[r] = *(const float4*)&Ht[(size_t)(m0 + row) * HID + k0 + c8];
                fa1[r] = *(const float4*)&Ht[(size_t)(m0 + row) * HID + k0 + c8 + 4];
                fb0[r] = *(const float4*)&Vw[(size_t)(n0 + row) * HID + k0 + c8];
                fb1[r] = *(const float4*)&Vw[(size_t)(n0 + row) * HID + k0 + c8 + 4];
            }
        }

        // ---- (2) compute: fragments are packed bf16, pure ds_load_b128
        Frag bfrag[2];
#pragma unroll
        for (int j = 0; j < 2; ++j) {
            // B 32x16: lane ll = col n, elements 0..15 = K hi*16 + 0..15
            const __bf16* brow = &Bs[cur][(nw * 32 + j * 16 + ll) * LSTR + hi * 16];
            bfrag[j].h[0] = *(const v8bf*)&brow[0];
            bfrag[j].h[1] = *(const v8bf*)&brow[8];
        }
#pragma unroll
        for (int i = 0; i < 4; ++i) {
            // A 16x32: lane ll = row m; elems 0..15 = K hi*16 + 0..15
            const __bf16* arow = &As[cur][(mw * 64 + i * 16 + ll) * LSTR];
            Frag af;
            af.h[0] = *(const v8bf*)&arow[hi * 16];
            af.h[1] = *(const v8bf*)&arow[hi * 16 + 8];
#pragma unroll
            for (int j = 0; j < 2; ++j) {
                acc[i][j] = __builtin_amdgcn_wmma_f32_16x16x32_bf16(
                    false, af.v, false, bfrag[j].v, (short)0, acc[i][j], false, false);
            }
        }

        // ---- (3) convert once + store next chunk into the other buffer
        if (kk < HID / KSTEP - 1) {
            const int nxt = cur ^ 1;
#pragma unroll
            for (int r = 0; r < 2; ++r) {
                int row = r ? srow1 : srow0;
                int c8  = r ? sc81  : sc80;
                *(v8bf*)&As[nxt][row * LSTR + c8] = cvt8(fa0[r], fa1[r]);
                *(v8bf*)&Bs[nxt][row * LSTR + c8] = cvt8(fb0[r], fb1[r]);
            }
        }
        // ---- (4) one barrier per K-step
        __syncthreads();
    }

    // ---- epilogue: C/D 16x16 layout: lane ll = N, VGPR r -> M = r + hi*8
#pragma unroll
    for (int j = 0; j < 2; ++j) {
        const int n = n0 + nw * 32 + j * 16 + ll;
        const float vb = Vb[n];
#pragma unroll
        for (int i = 0; i < 4; ++i) {
            const int mbase = m0 + mw * 64 + i * 16 + hi * 8;
#pragma unroll
            for (int r = 0; r < 8; ++r) {
                logits[(size_t)(mbase + r) * VOCAB + n] = acc[i][j][r] + vb;
            }
        }
    }
}

// ---------------------------------------------------------------------------
extern "C" void kernel_launch(void* const* d_in, const int* in_sizes, int n_in,
                              void* d_out, int out_size, void* d_ws, size_t ws_size,
                              hipStream_t stream)
{
    (void)in_sizes; (void)n_in; (void)out_size; (void)d_ws; (void)ws_size;

    const int*   x  = (const int*)  d_in[0];
    const int*   z  = (const int*)  d_in[1];
    const float* h  = (const float*)d_in[2];
    const float* E  = (const float*)d_in[3];
    const float* Vw = (const float*)d_in[4];
    const float* Vb = (const float*)d_in[5];
    const float* W  = (const float*)d_in[6];
    const float* U  = (const float*)d_in[7];

    float* logits = (float*)d_out;                       // (2048, 1, 32000)
    float* ht     = logits + (size_t)BATCH * VOCAB;      // (2048, 1, 256)

    rnn_step_kernel<<<BATCH, HID, 0, stream>>>(x, z, h, E, W, U, ht);

    dim3 grid(VOCAB / NT, BATCH / MT);                   // 250 x 16
    logits_wmma_kernel<<<grid, 256, 0, stream>>>(ht, Vw, Vb, logits);
}
